// HierEncoder_85083302134411
// MI455X (gfx1250) — matrix operand order repaired and checked by reading, not compiled, not run on hardware
//
#include <hip/hip_runtime.h>
#include <math.h>

// ---------------------------------------------------------------------------
// GAT forward for MI455X (gfx1250), wave32 + WMMA bf16 16x16x32.
// Flash-style softmax: attention [3,8192,8192] is never materialized.
// Main kernel register-blocks 2 row tiles per wave to halve L2 traffic on
// the streamed B operand (WhT, L2-resident at 34.6 MB << 192 MB L2).
// ---------------------------------------------------------------------------

#define NN 8192
#define FF 690
#define FP 704          // 690 padded to 44*16
#define HH 3
#define LRELU 0.2f

typedef __attribute__((ext_vector_type(16))) __bf16 v16bf;
typedef __attribute__((ext_vector_type(8)))  float  v8f;

union Frag {
  v16bf  v;
  __bf16 e[16];
  uint4  q[2];
};

// ---------------------------------------------------------------------------
// K0a: x [N,690] f32 -> xb [N,704] bf16 (zero padded)
// ---------------------------------------------------------------------------
__global__ __launch_bounds__(256) void cvt_x(const float* __restrict__ x,
                                             __bf16* __restrict__ xb) {
  int idx = blockIdx.x * 256 + threadIdx.x;
  if (idx >= NN * FP) return;
  int n = idx / FP, k = idx % FP;
  xb[idx] = (k < FF) ? (__bf16)x[(size_t)n * FF + k] : (__bf16)0.0f;
}

// ---------------------------------------------------------------------------
// K0b: W [H,690,690] f32 -> wt [H,704(o),704(k)] bf16 transposed, zero padded
// ---------------------------------------------------------------------------
__global__ __launch_bounds__(256) void cvt_w(const float* __restrict__ W,
                                             __bf16* __restrict__ wt) {
  int idx = blockIdx.x * 256 + threadIdx.x;
  if (idx >= HH * FP * FP) return;
  int h = idx / (FP * FP);
  int r = idx % (FP * FP);
  int o = r / FP, k = r % FP;
  wt[idx] = (o < FF && k < FF) ? (__bf16)W[((size_t)h * FF + k) * FF + o]
                               : (__bf16)0.0f;
}

// ---------------------------------------------------------------------------
// K1: Wh = x @ W[h] via WMMA, output transposed bf16: whT[h][o][n]
// One wave per 16x16 output tile; K loop over 704 in steps of 32.
// ---------------------------------------------------------------------------
__global__ __launch_bounds__(256) void gat_gemm1(const __bf16* __restrict__ xb,
                                                 const __bf16* __restrict__ wt,
                                                 __bf16* __restrict__ whT) {
  const int w    = threadIdx.x >> 5;
  const int lane = threadIdx.x & 31;
  const int m    = lane & 15;
  const int hi   = lane >> 4;

  const int ctiles = FP / 16;                 // 44
  const int rtiles = NN / 16;                 // 512
  int tile = blockIdx.x * 8 + w;              // 67584 tiles total
  int h  = tile / (rtiles * ctiles);
  int r  = tile % (rtiles * ctiles);
  int rb = r / ctiles;
  int ct = r % ctiles;
  const int n0 = rb * 16;
  const int o0 = ct * 16;

  // A: row = lane&15, K runs {8*hi..+7, 16+8*hi..+7}
  const __bf16* arow = xb + (size_t)(n0 + m) * FP + 8 * hi;
  // B: col = lane&15, K runs {8*hi..+7, 16+8*hi..+7}
  const __bf16* brow = wt + ((size_t)h * FP + (o0 + m)) * FP + 8 * hi;

  v8f acc = {};
  #pragma unroll 2
  for (int k = 0; k < FP; k += 32) {
    Frag A, B;
    A.q[0] = *(const uint4*)(arow + k);
    A.q[1] = *(const uint4*)(arow + k + 16);
    B.q[0] = *(const uint4*)(brow + k);
    B.q[1] = *(const uint4*)(brow + k + 16);
    acc = __builtin_amdgcn_wmma_f32_16x16x32_bf16(false, A.v, false, B.v,
                                                  (short)0, acc, false, false);
  }

  // C: row = v + 8*hi, col = lane&15 ; store transposed (8 consecutive bf16)
  __bf16* out = whT + ((size_t)h * FP + (o0 + m)) * NN + n0 + 8 * hi;
  #pragma unroll
  for (int v = 0; v < 8; ++v) out[v] = (__bf16)acc[v];
}

// ---------------------------------------------------------------------------
// K2: a1[h,n] = Wh[h,n,:].a[h,:F] ; a2[h,n] = Wh[h,n,:].a[h,F:]
// Coalesced column reads of whT across threads.
// ---------------------------------------------------------------------------
__global__ __launch_bounds__(256) void gat_a12(const __bf16* __restrict__ whT,
                                               const float* __restrict__ a,
                                               float* __restrict__ a1,
                                               float* __restrict__ a2) {
  int id = blockIdx.x * 256 + threadIdx.x;
  if (id >= HH * NN) return;
  int h = id / NN, n = id % NN;
  const __bf16* base = whT + (size_t)h * FP * NN + n;
  const float*  av   = a + h * 2 * FF;
  float s1 = 0.0f, s2 = 0.0f;
  for (int o = 0; o < FF; ++o) {
    float wv = (float)base[(size_t)o * NN];
    s1 += wv * av[o];
    s2 += wv * av[FF + o];
  }
  a1[id] = s1;
  a2[id] = s2;
}

// ---------------------------------------------------------------------------
// K3: per-(head,row) online softmax stats (max m, denom l) over masked scores.
// One wave per row; 3 heads per pass so adj is streamed ONCE (256 MB).
// ---------------------------------------------------------------------------
__global__ __launch_bounds__(256) void gat_stats(const int* __restrict__ adj,
                                                 const float* __restrict__ a1,
                                                 const float* __restrict__ a2,
                                                 float* __restrict__ mstat,
                                                 float* __restrict__ lstat) {
  const int w    = threadIdx.x >> 5;
  const int lane = threadIdx.x & 31;
  const int i    = blockIdx.x * 8 + w;

  float a1v[HH], mh[HH], lh[HH];
  #pragma unroll
  for (int h = 0; h < HH; ++h) {
    a1v[h] = a1[h * NN + i];
    mh[h]  = -3.0e38f;
    lh[h]  = 0.0f;
  }
  const int* arow = adj + (size_t)i * NN + lane * 4;

  for (int t = 0; t < NN / 128; ++t) {
    int aArr[4];
    *(int4*)aArr = *(const int4*)(arow + t * 128);
    float a2l[HH][4];
    #pragma unroll
    for (int h = 0; h < HH; ++h)
      *(float4*)&a2l[h][0] = *(const float4*)(a2 + h * NN + t * 128 + lane * 4);
    #pragma unroll
    for (int c = 0; c < 4; ++c) {
      #pragma unroll
      for (int h = 0; h < HH; ++h) {
        if (aArr[c] > 0) {
          float s = a1v[h] + a2l[h][c];
          float e = s > 0.0f ? s : LRELU * s;
          if (e > mh[h]) {
            lh[h] = lh[h] * __expf(mh[h] - e) + 1.0f;
            mh[h] = e;
          } else {
            lh[h] += __expf(e - mh[h]);
          }
        }
      }
    }
  }
  // cross-lane (wave32) combine of (m,l)
  #pragma unroll
  for (int h = 0; h < HH; ++h) {
    float m = mh[h], l = lh[h];
    for (int off = 16; off > 0; off >>= 1) {
      float m2 = __shfl_xor(m, off, 32);
      float l2 = __shfl_xor(l, off, 32);
      float nm = fmaxf(m, m2);
      l = l * __expf(m - nm) + l2 * __expf(m2 - nm);
      m = nm;
    }
    if (lane == 0) { mstat[h * NN + i] = m; lstat[h * NN + i] = l; }
  }
}

// ---------------------------------------------------------------------------
// K4: fused  y[i,:] = elu( mean_h( softmax(mask(e_h[i,:])) @ Wh_h ) )
// Workgroup = 12 waves = 3 heads x 4 column groups, owns a 32-row block
// (2 row tiles register-blocked per wave -> each B fragment feeds 2 WMMAs,
// halving L2 traffic on the streamed WhT operand).
// Producer waves (g<2) build the bf16 probability A-fragment for row tile g
// of head h in LDS each k-step. 1/l folded into epilogue; heads reduced via
// ds_add_f32 in LDS, two passes over row halves to stay within 64KB LDS.
// ---------------------------------------------------------------------------
__global__ __launch_bounds__(384) void gat_attn(const int* __restrict__ adj,
                                                const __bf16* __restrict__ whT,
                                                const float* __restrict__ a1,
                                                const float* __restrict__ a2,
                                                const float* __restrict__ mstat,
                                                const float* __restrict__ lstat,
                                                float* __restrict__ y) {
  __shared__ alignas(16) __bf16 ldsA[HH][2][32][16];  // 6 KB
  __shared__ float red[16][FP];                       // 44 KB

  const int w    = threadIdx.x >> 5;
  const int lane = threadIdx.x & 31;
  const int m    = lane & 15;
  const int hi   = lane >> 4;
  const int h    = w % 3;     // head
  const int g    = w / 3;     // column group (0..3), 176 cols each
  const int i0   = blockIdx.x * 32;

  // producer-side per-lane constants: wave (h, g<2) produces row tile rt=g
  const int    prt    = (g < 2) ? g : 0;
  const int    prow   = i0 + prt * 16 + m;
  const float  a1v    = a1[h * NN + prow];
  const float  mrow   = mstat[h * NN + prow];
  const int*   adjrow = adj + (size_t)prow * NN + 8 * hi;
  const float* a2row  = a2 + h * NN + 8 * hi;

  // epilogue scale 1/l per C row (row = v + 8*hi) for both row tiles
  float invl[2][8];
  #pragma unroll
  for (int rt = 0; rt < 2; ++rt)
    #pragma unroll
    for (int v = 0; v < 8; ++v)
      invl[rt][v] = 1.0f / lstat[h * NN + i0 + rt * 16 + v + 8 * hi];

  // B fragment base (col = lane&15 within tile 0 of this wave's strip);
  // tile t adds a compile-time offset t*16*NN that folds into the
  // instruction's immediate offset.
  const __bf16* Bbase = whT + ((size_t)h * FP + g * 176 + m) * NN + 8 * hi;

  v8f acc[2][11];
  #pragma unroll
  for (int rt = 0; rt < 2; ++rt)
    #pragma unroll
    for (int t = 0; t < 11; ++t) acc[rt][t] = (v8f){};

  for (int j = 0; j < NN; j += 32) {
    if (g < 2) {  // 6 producer waves: (head h, row tile g)
      __builtin_prefetch(adjrow + j + 32, 0, 0);  // next adj block -> L2
      int   adjv[16];
      float a2l[16];
      *(int4*)&adjv[0]   = *(const int4*)(adjrow + j);
      *(int4*)&adjv[4]   = *(const int4*)(adjrow + j + 4);
      *(int4*)&adjv[8]   = *(const int4*)(adjrow + j + 16);
      *(int4*)&adjv[12]  = *(const int4*)(adjrow + j + 20);
      *(float4*)&a2l[0]  = *(const float4*)(a2row + j);
      *(float4*)&a2l[4]  = *(const float4*)(a2row + j + 4);
      *(float4*)&a2l[8]  = *(const float4*)(a2row + j + 16);
      *(float4*)&a2l[12] = *(const float4*)(a2row + j + 20);
      Frag P;
      #pragma unroll
      for (int q = 0; q < 16; ++q) {
        float s = a1v + a2l[q];
        float e = s > 0.0f ? s : LRELU * s;
        float p = (adjv[q] > 0) ? __expf(e - mrow) : 0.0f;
        P.e[q] = (__bf16)p;
      }
      *(uint4*)&ldsA[h][prt][lane][0] = P.q[0];
      *(uint4*)&ldsA[h][prt][lane][8] = P.q[1];
    }
    __syncthreads();

    Frag A0, A1;
    A0.q[0] = *(const uint4*)&ldsA[h][0][lane][0];
    A0.q[1] = *(const uint4*)&ldsA[h][0][lane][8];
    A1.q[0] = *(const uint4*)&ldsA[h][1][lane][0];
    A1.q[1] = *(const uint4*)&ldsA[h][1][lane][8];
    #pragma unroll
    for (int t = 0; t < 11; ++t) {
      Frag B;
      B.q[0] = *(const uint4*)(Bbase + (size_t)t * 16 * NN + j);
      B.q[1] = *(const uint4*)(Bbase + (size_t)t * 16 * NN + j + 16);
      acc[0][t] = __builtin_amdgcn_wmma_f32_16x16x32_bf16(false, A0.v, false,
                                                          B.v, (short)0,
                                                          acc[0][t],
                                                          false, false);
      acc[1][t] = __builtin_amdgcn_wmma_f32_16x16x32_bf16(false, A1.v, false,
                                                          B.v, (short)0,
                                                          acc[1][t],
                                                          false, false);
    }
    __syncthreads();
  }

  // epilogue: two passes over row halves (reuse one 16x704 LDS buffer)
  #pragma unroll
  for (int rt = 0; rt < 2; ++rt) {
    for (int idx = threadIdx.x; idx < 16 * FP; idx += 384)
      (&red[0][0])[idx] = 0.0f;
    __syncthreads();

    #pragma unroll
    for (int t = 0; t < 11; ++t) {
      const int col = g * 176 + t * 16 + m;
      #pragma unroll
      for (int v = 0; v < 8; ++v)
        atomicAdd(&red[v + 8 * hi][col], acc[rt][t][v] * invl[rt][v]);
    }
    __syncthreads();

    for (int idx = threadIdx.x; idx < 16 * FP; idx += 384) {
      int r = idx / FP, c = idx % FP;
      if (c < FF) {
        float s = red[r][c] * (1.0f / 3.0f);
        float o = s > 0.0f ? s : (__expf(s) - 1.0f);
        y[(size_t)(i0 + rt * 16 + r) * FF + c] = o;
      }
    }
    __syncthreads();
  }
}

// ---------------------------------------------------------------------------
extern "C" void kernel_launch(void* const* d_in, const int* in_sizes, int n_in,
                              void* d_out, int out_size, void* d_ws,
                              size_t ws_size, hipStream_t stream) {
  (void)in_sizes; (void)n_in; (void)out_size; (void)ws_size;

  const float* x   = (const float*)d_in[0];   // [8192,690] f32
  const int*   adj = (const int*)d_in[1];     // [8192,8192] i32
  const float* W   = (const float*)d_in[2];   // [3,690,690] f32
  const float* a   = (const float*)d_in[3];   // [3,1380,1] f32
  float*       y   = (float*)d_out;           // [8192,690] f32

  char* ws  = (char*)d_ws;
  size_t off = 0;
  auto alloc = [&](size_t bytes) -> char* {
    char* p = ws + off;
    off += (bytes + 255) & ~(size_t)255;
    return p;
  };
  __bf16* xb  = (__bf16*)alloc((size_t)NN * FP * 2);        // 11.5 MB
  __bf16* wt  = (__bf16*)alloc((size_t)HH * FP * FP * 2);   //  3.0 MB
  __bf16* whT = (__bf16*)alloc((size_t)HH * FP * NN * 2);   // 34.6 MB (L2-resident)
  float*  a1  = (float*)alloc((size_t)HH * NN * 4);
  float*  a2  = (float*)alloc((size_t)HH * NN * 4);
  float*  ms  = (float*)alloc((size_t)HH * NN * 4);
  float*  ls  = (float*)alloc((size_t)HH * NN * 4);

  { int n = NN * FP;      cvt_x<<<(n + 255) / 256, 256, 0, stream>>>(x, xb); }
  { int n = HH * FP * FP; cvt_w<<<(n + 255) / 256, 256, 0, stream>>>(W, wt); }

  gat_gemm1<<<(HH * (NN / 16) * (FP / 16)) / 8, 256, 0, stream>>>(xb, wt, whT);
  gat_a12<<<(HH * NN + 255) / 256, 256, 0, stream>>>(whT, a, a1, a2);
  gat_stats<<<NN / 8, 256, 0, stream>>>(adj, a1, a2, ms, ls);
  gat_attn<<<NN / 32, 384, 0, stream>>>(adj, whT, a1, a2, ms, ls, y);
}
